// DynamicGAE_84224308674817
// MI455X (gfx1250) — matrix operand order repaired and checked by reading, not compiled, not run on hardware
//
#include <hip/hip_runtime.h>

typedef unsigned short u16;
typedef __attribute__((ext_vector_type(8)))  unsigned short u16x8;
typedef __attribute__((ext_vector_type(16))) __bf16 v16bf;
typedef __attribute__((ext_vector_type(8)))  float v8f;

// ---------------------------------------------------------------- helpers

__device__ __forceinline__ u16 bf16rne(float f) {
    unsigned u = __float_as_uint(f);
    unsigned r = u + 0x7FFFu + ((u >> 16) & 1u);   // round-to-nearest-even
    return (u16)(r >> 16);
}

__device__ __forceinline__ void atomicMaxF(float* a, float v) {
    // sign-split trick: positive floats order as ints, negatives reversed as unsigned
    if (v >= 0.0f) atomicMax((int*)a, __float_as_int(v));
    else           atomicMin((unsigned int*)a, __float_as_uint(v));
}

// ---------------------------------------------------------------- converts

__global__ void k_f2bf(const float* __restrict__ s, u16* __restrict__ d, long long n) {
    long long i = (long long)blockIdx.x * blockDim.x + threadIdx.x;
    if (i < n) d[i] = bf16rne(s[i]);
}

// W [R x C] row-major f32  ->  WT [C x R] row-major bf16
__global__ void k_f2bfT(const float* __restrict__ s, u16* __restrict__ d, int R, int C) {
    int i = blockIdx.x * blockDim.x + threadIdx.x;
    if (i >= R * C) return;
    int r = i / C, c = i % C;
    d[(long long)c * R + r] = bf16rne(s[i]);
}

__global__ void k_fill(float* __restrict__ p, float v, long long n) {
    long long i = (long long)blockIdx.x * blockDim.x + threadIdx.x;
    if (i < n) p[i] = v;
}

// ---------------------------------------------------------------- WMMA GEMM
// One wave32 computes a full 16 x Nc row-strip (NT = Nc/16 tiles); the A fragment
// is loaded once per k-step and reused across NT WMMAs. The weight matrix
// WT [Nc x KK] (bf16) is staged in LDS once per workgroup and B fragments are
// read with ds_load_b128 (DScnt), decoupled from the global A stream (LOADcnt).
// All NT B fragments are preloaded into registers so the NT WMMAs issue
// back-to-back behind a single wait.
// Epilogue: optional bias[col], optional BatchNorm(col), act 0/1/2 = none/relu/elu.
// Outputs: optional f32 (Cout) and/or bf16 (Cbf).
template<int NT, int KK>
__global__ __launch_bounds__(256)
void k_gemm_wmma(const u16* __restrict__ A, const u16* __restrict__ WT,
                 float* __restrict__ Cout, u16* __restrict__ Cbf,
                 int M,
                 const float* __restrict__ bias,
                 const float* __restrict__ bng, const float* __restrict__ bnb,
                 const float* __restrict__ bnm, const float* __restrict__ bnv,
                 int act)
{
    constexpr int Nc = NT * 16;
    __shared__ u16 sB[Nc * KK];

    // ---- cooperative stage of WT into LDS (b128 chunks)
    const int tid = threadIdx.x;
    constexpr int chunks = (Nc * KK) / 8;          // # of 16-byte chunks
    #pragma unroll
    for (int i = 0; i < chunks / 256; ++i)
        ((u16x8*)sB)[tid + i * 256] = ((const u16x8*)WT)[tid + i * 256];
    if constexpr (chunks % 256) {
        int i = (chunks / 256) * 256 + tid;
        if (tid < (chunks % 256)) ((u16x8*)sB)[i] = ((const u16x8*)WT)[i];
    }
    __syncthreads();

    const int lane = tid & 31;
    const int wid  = blockIdx.x * 8 + (tid >> 5);
    if (wid >= (M >> 4)) return;       // wave-uniform: EXEC all-1s inside

    const int m0  = wid << 4;
    const int row = lane & 15;
    const int kg  = lane >> 4;         // 0 for lanes 0-15, 1 for lanes 16-31

    // A frag (16x32, 16-bit): lanes 0-15 carry K [0..7]+[16..23], lanes 16-31 [8..15]+[24..31]
    const u16* arow = A + (long long)(m0 + row) * KK + kg * 8;

    v8f acc[NT];
    #pragma unroll
    for (int t = 0; t < NT; ++t) acc[t] = (v8f){};

    #pragma unroll
    for (int k0 = 0; k0 < KK; k0 += 32) {
        union Frag { u16x8 u[2]; v16bf v; };
        Frag fa;
        fa.u[0] = *(const u16x8*)(arow + k0);
        fa.u[1] = *(const u16x8*)(arow + k0 + 16);

        // B frag (32x16): lane = column; lanes 0-15 carry K [0..7]+[16..23], lanes 16-31 [8..15]+[24..31]
        Frag fb[NT];
        #pragma unroll
        for (int t = 0; t < NT; ++t) {
            const u16* brow = sB + (t * 16 + row) * KK + kg * 8 + k0;
            fb[t].u[0] = *(const u16x8*)(brow);
            fb[t].u[1] = *(const u16x8*)(brow + 16);
        }
        #pragma unroll
        for (int t = 0; t < NT; ++t)
            acc[t] = __builtin_amdgcn_wmma_f32_16x16x32_bf16(false, fa.v, false, fb[t].v,
                                                             (short)0, acc[t], false, false);
    }

    // C/D layout: lanes 0-15 -> M = m0 + r, lanes 16-31 -> M = m0 + 8 + r; N = tile*16 + (lane&15)
    const int mbase = m0 + kg * 8;
    #pragma unroll
    for (int t = 0; t < NT; ++t) {
        const int ncol = t * 16 + row;
        const float bc = bias ? bias[ncol] : 0.0f;
        float g = 1.f, bb = 0.f, mm = 0.f, vv = 1.f;
        if (bng) { g = bng[ncol]; bb = bnb[ncol]; mm = bnm[ncol]; vv = bnv[ncol]; }
        #pragma unroll
        for (int r = 0; r < 8; ++r) {
            float x = acc[t][r] + bc;
            if (bng) x = (x - mm) * rsqrtf(vv + 1e-5f) * g + bb;
            if (act == 1)      x = x > 0.f ? x : 0.f;
            else if (act == 2) x = x > 0.f ? x : (__expf(x) - 1.f);
            const long long idx = (long long)(mbase + r) * Nc + ncol;
            if (Cout) Cout[idx] = x;
            if (Cbf)  Cbf[idx]  = bf16rne(x);
        }
    }
}

// ---------------------------------------------------------------- attention

// a_src[n,h] = dot(h1[n,h,:], att_src[h,:]); same for dst.
__global__ void k_att_dots(const float* __restrict__ h,
                           const float* __restrict__ atts, const float* __restrict__ attd,
                           float* __restrict__ as_, float* __restrict__ ad_,
                           int Nn, int H, int Cc)
{
    int i = blockIdx.x * blockDim.x + threadIdx.x;
    if (i >= Nn * H) return;
    int n = i / H, hh = i % H;
    const float* hp = h + (long long)n * H * Cc + hh * Cc;
    const float* sp = atts + hh * Cc;
    const float* dp = attd + hh * Cc;
    float s0 = 0.f, s1 = 0.f;
    for (int c = 0; c < Cc; ++c) { float hv = hp[c]; s0 += hv * sp[c]; s1 += hv * dp[c]; }
    as_[i] = s0; ad_[i] = s1;
}

// segment max of leaky_relu(a_src[src]+a_dst[dst]) over incoming edges (+self loops)
__global__ void k_edge_max(const int* __restrict__ src, const int* __restrict__ dst,
                           int E_, int Nn, int H,
                           const float* __restrict__ as_, const float* __restrict__ ad_,
                           float* __restrict__ emax)
{
    long long idx = (long long)blockIdx.x * blockDim.x + threadIdx.x;
    long long tot = (long long)(E_ + Nn) * H;
    if (idx >= tot) return;
    int e = (int)(idx / H), hh = (int)(idx % H);
    int s = e < E_ ? src[e] : e - E_;
    int d = e < E_ ? dst[e] : e - E_;
    float v = as_[s * H + hh] + ad_[d * H + hh];
    v = v > 0.f ? v : 0.2f * v;
    atomicMaxF(&emax[d * H + hh], v);
}

__global__ void k_edge_sum(const int* __restrict__ src, const int* __restrict__ dst,
                           int E_, int Nn, int H,
                           const float* __restrict__ as_, const float* __restrict__ ad_,
                           const float* __restrict__ emax, float* __restrict__ den)
{
    long long idx = (long long)blockIdx.x * blockDim.x + threadIdx.x;
    long long tot = (long long)(E_ + Nn) * H;
    if (idx >= tot) return;
    int e = (int)(idx / H), hh = (int)(idx % H);
    int s = e < E_ ? src[e] : e - E_;
    int d = e < E_ ? dst[e] : e - E_;
    float v = as_[s * H + hh] + ad_[d * H + hh];
    v = v > 0.f ? v : 0.2f * v;
    atomicAdd(&den[d * H + hh], __expf(v - emax[d * H + hh]));
}

// agg[dst, h*Cc+c] += h1[src, h*Cc+c] * alpha(edge,h)
__global__ void k_edge_agg(const int* __restrict__ src, const int* __restrict__ dst,
                           int E_, int Nn, int H, int Cc,
                           const float* __restrict__ as_, const float* __restrict__ ad_,
                           const float* __restrict__ emax, const float* __restrict__ den,
                           const float* __restrict__ hsrc, float* __restrict__ agg)
{
    long long idx = (long long)blockIdx.x * blockDim.x + threadIdx.x;
    const int HC = H * Cc;
    long long tot = (long long)(E_ + Nn) * HC;
    if (idx >= tot) return;
    int e = (int)(idx / HC);
    int c = (int)(idx % HC);
    int hh = c / Cc;
    int s = e < E_ ? src[e] : e - E_;
    int d = e < E_ ? dst[e] : e - E_;
    float v = as_[s * H + hh] + ad_[d * H + hh];
    v = v > 0.f ? v : 0.2f * v;
    float alpha = __expf(v - emax[d * H + hh]) / (den[d * H + hh] + 1e-16f);
    atomicAdd(&agg[(long long)d * HC + c], hsrc[(long long)s * HC + c] * alpha);
}

// x = act( BN( x + bias ) ), per column; act 0/1/2 = none/relu/elu; BN optional;
// optional bf16 mirror
__global__ void k_bias_bn_act(float* __restrict__ x, u16* __restrict__ xb,
                              const float* __restrict__ bias,
                              const float* __restrict__ g, const float* __restrict__ b,
                              const float* __restrict__ m, const float* __restrict__ v,
                              int Nn, int Cc, int act)
{
    long long i = (long long)blockIdx.x * blockDim.x + threadIdx.x;
    if (i >= (long long)Nn * Cc) return;
    int c = (int)(i % Cc);
    float t = x[i] + (bias ? bias[c] : 0.f);
    if (g) t = (t - m[c]) * rsqrtf(v[c] + 1e-5f) * g[c] + b[c];
    if (act == 2)      t = t > 0.f ? t : (__expf(t) - 1.f);
    else if (act == 1) t = t > 0.f ? t : 0.f;
    x[i] = t;
    if (xb) xb[i] = bf16rne(t);
}

// ---------------------------------------------------------------- launch

static inline int divUp(long long a, int b) { return (int)((a + (long long)b - 1) / b); }

extern "C" void kernel_launch(void* const* d_in, const int* in_sizes, int n_in,
                              void* d_out, int out_size, void* d_ws, size_t ws_size,
                              hipStream_t stream)
{
    const int IN_CH = 128, HIDC = 32, H1 = 4, OUTC = 64;
    const int Nn = in_sizes[0] / IN_CH;
    const int E_ = in_sizes[n_in - 1] / 2;

    // ---- input pointers (setup_inputs dict order, depth-first)
    const float* x      = (const float*)d_in[0];
    const float* c1W    = (const float*)d_in[1];
    const float* c1as   = (const float*)d_in[2];
    const float* c1ad   = (const float*)d_in[3];
    const float* c1b    = (const float*)d_in[4];
    const float* bn1g   = (const float*)d_in[5];
    const float* bn1b   = (const float*)d_in[6];
    const float* bn1m   = (const float*)d_in[7];
    const float* bn1v   = (const float*)d_in[8];
    const float* c2W    = (const float*)d_in[9];
    const float* c2as   = (const float*)d_in[10];
    const float* c2ad   = (const float*)d_in[11];
    const float* c2b    = (const float*)d_in[12];
    const float* dict   = (const float*)d_in[13];
    // p_1 / p_2 / p_12 : {W1,b1,bn_g,bn_b,bn_m,bn_v,W2,b2} starting at 14, 22, 30
    const int pbase[3] = {14, 22, 30};
    const int pout[3]  = {64, 64, 128};
    const int* ei   = (const int*)d_in[n_in - 1];
    const int* esrc = ei;
    const int* edst = ei + E_;
    float* out = (float*)d_out;

    // ---- workspace layout
    char* ws = (char*)d_ws;
    size_t off = 0;
    auto alloc = [&](size_t bytes) -> void* {
        void* p = ws + off;
        off += (bytes + 255) & ~(size_t)255;
        return p;
    };
    u16* xb     = (u16*)alloc((size_t)Nn * 128 * 2);   // x in bf16
    u16* wt_c1  = (u16*)alloc(128 * 128 * 2);          // conv1 W^T
    u16* wt_c2  = (u16*)alloc(64 * 128 * 2);           // conv2 W^T
    u16* wt_d1  = (u16*)alloc(128 * 128 * 2);          // D^T   (for x@D)
    u16* wt_d2  = (u16*)alloc(128 * 128 * 2);          // D     (for zs@D^T)
    u16* wt_a[3]; u16* wt_b[3];
    for (int p = 0; p < 3; ++p) {
        wt_a[p] = (u16*)alloc(128 * 64 * 2);           // W1^T [128 x 64]
        wt_b[p] = (u16*)alloc((size_t)pout[p] * 128 * 2); // W2^T [out x 128]
    }
    float* h1   = (float*)alloc((size_t)Nn * 128 * 4); // conv1 linear out
    float* agg1 = (float*)alloc((size_t)Nn * 128 * 4); // conv1 aggregated -> h
    float* as1  = (float*)alloc((size_t)Nn * 4 * 4);
    float* ad1  = (float*)alloc((size_t)Nn * 4 * 4);
    float* emx1 = (float*)alloc((size_t)Nn * 4 * 4);
    float* den1 = (float*)alloc((size_t)Nn * 4 * 4);
    u16*   hb   = (u16*)alloc((size_t)Nn * 128 * 2);   // h bf16 (conv2 input)
    float* h2   = (float*)alloc((size_t)Nn * 64 * 4);
    float* agg2 = (float*)alloc((size_t)Nn * 64 * 4);  // -> z
    u16*   zb   = (u16*)alloc((size_t)Nn * 64 * 2);    // z bf16
    u16*   mhb  = (u16*)alloc((size_t)Nn * 128 * 2);   // MLP hidden / zs (bf16 only)
    (void)ws_size; (void)out_size;

    const int B = 256;
    auto gemm = [&](const u16* A, const u16* WT, float* C, u16* Cb,
                    int M, int K, int Ncol,
                    const float* bias, const float* g, const float* bb,
                    const float* mm, const float* vv, int act) {
        int grid = divUp(M / 16, 8);
        if (Ncol == 128 && K == 128)
            k_gemm_wmma<8, 128><<<grid, B, 0, stream>>>(A, WT, C, Cb, M, bias, g, bb, mm, vv, act);
        else if (Ncol == 64 && K == 128)
            k_gemm_wmma<4, 128><<<grid, B, 0, stream>>>(A, WT, C, Cb, M, bias, g, bb, mm, vv, act);
        else // Ncol == 128, K == 64
            k_gemm_wmma<8, 64><<<grid, B, 0, stream>>>(A, WT, C, Cb, M, bias, g, bb, mm, vv, act);
    };

    // ---- weight/activation conversions
    k_f2bf <<<divUp((long long)Nn * 128, B), B, 0, stream>>>(x, xb, (long long)Nn * 128);
    k_f2bfT<<<divUp(128 * 128, B), B, 0, stream>>>(c1W, wt_c1, 128, 128);
    k_f2bfT<<<divUp(128 * 64,  B), B, 0, stream>>>(c2W, wt_c2, 128, 64);
    k_f2bfT<<<divUp(128 * 128, B), B, 0, stream>>>(dict, wt_d1, 128, 128);
    k_f2bf <<<divUp(128 * 128, B), B, 0, stream>>>(dict, wt_d2, 128 * 128);
    for (int p = 0; p < 3; ++p) {
        k_f2bfT<<<divUp(64 * 128, B), B, 0, stream>>>((const float*)d_in[pbase[p] + 0], wt_a[p], 64, 128);
        k_f2bfT<<<divUp(128 * pout[p], B), B, 0, stream>>>((const float*)d_in[pbase[p] + 6], wt_b[p], 128, pout[p]);
    }

    const long long EEH1 = (long long)(E_ + Nn) * H1;
    const long long EEH2 = (long long)(E_ + Nn);

    // ---- conv1: h1 = x @ W  (WMMA), attention softmax, aggregate
    gemm(xb, wt_c1, h1, nullptr, Nn, 128, 128, nullptr, nullptr, nullptr, nullptr, nullptr, 0);
    k_att_dots<<<divUp(Nn * H1, B), B, 0, stream>>>(h1, c1as, c1ad, as1, ad1, Nn, H1, HIDC);
    k_fill<<<divUp((long long)Nn * H1, B), B, 0, stream>>>(emx1, -1e30f, (long long)Nn * H1);
    k_fill<<<divUp((long long)Nn * H1, B), B, 0, stream>>>(den1, 0.f, (long long)Nn * H1);
    k_fill<<<divUp((long long)Nn * 128, B), B, 0, stream>>>(agg1, 0.f, (long long)Nn * 128);
    k_edge_max<<<divUp(EEH1, B), B, 0, stream>>>(esrc, edst, E_, Nn, H1, as1, ad1, emx1);
    k_edge_sum<<<divUp(EEH1, B), B, 0, stream>>>(esrc, edst, E_, Nn, H1, as1, ad1, emx1, den1);
    k_edge_agg<<<divUp(EEH1 * 32, B), B, 0, stream>>>(esrc, edst, E_, Nn, H1, HIDC,
                                                      as1, ad1, emx1, den1, h1, agg1);
    // + conv1 bias, BN(bn1), ELU  -> h (f32 + bf16 in one pass)
    k_bias_bn_act<<<divUp((long long)Nn * 128, B), B, 0, stream>>>(agg1, hb, c1b,
                                                                   bn1g, bn1b, bn1m, bn1v,
                                                                   Nn, 128, 2);

    // ---- conv2 (1 head, 64 ch)
    gemm(hb, wt_c2, h2, nullptr, Nn, 128, 64, nullptr, nullptr, nullptr, nullptr, nullptr, 0);
    k_att_dots<<<divUp(Nn, B), B, 0, stream>>>(h2, c2as, c2ad, as1, ad1, Nn, 1, OUTC);
    k_fill<<<divUp((long long)Nn, B), B, 0, stream>>>(emx1, -1e30f, (long long)Nn);
    k_fill<<<divUp((long long)Nn, B), B, 0, stream>>>(den1, 0.f, (long long)Nn);
    k_fill<<<divUp((long long)Nn * 64, B), B, 0, stream>>>(agg2, 0.f, (long long)Nn * 64);
    k_edge_max<<<divUp(EEH2, B), B, 0, stream>>>(esrc, edst, E_, Nn, 1, as1, ad1, emx1);
    k_edge_sum<<<divUp(EEH2, B), B, 0, stream>>>(esrc, edst, E_, Nn, 1, as1, ad1, emx1, den1);
    k_edge_agg<<<divUp(EEH2 * 64, B), B, 0, stream>>>(esrc, edst, E_, Nn, 1, OUTC,
                                                      as1, ad1, emx1, den1, h2, agg2);
    // z = elu(agg2 + bias)  (f32 + bf16 in one pass)
    k_bias_bn_act<<<divUp((long long)Nn * 64, B), B, 0, stream>>>(agg2, zb, c2b,
                                                                  nullptr, nullptr, nullptr, nullptr,
                                                                  Nn, 64, 2);

    // ---- dictionary: emb = relu(x@D) @ D^T   (both WMMA; intermediate kept bf16-only)
    gemm(xb, wt_d1, nullptr, mhb, Nn, 128, 128, nullptr, nullptr, nullptr, nullptr, nullptr, 1);
    gemm(mhb, wt_d2, out + (long long)Nn * 256, nullptr, Nn, 128, 128,
         nullptr, nullptr, nullptr, nullptr, nullptr, 0);

    // ---- prediction MLPs: relu(BN(z@W1+b1)) @ W2 + b2  (fused epilogues, bf16 hidden)
    long long ooff[3] = {0, (long long)Nn * 64, (long long)Nn * 128};
    for (int p = 0; p < 3; ++p) {
        const float* b1  = (const float*)d_in[pbase[p] + 1];
        const float* bg  = (const float*)d_in[pbase[p] + 2];
        const float* bb2 = (const float*)d_in[pbase[p] + 3];
        const float* bm  = (const float*)d_in[pbase[p] + 4];
        const float* bv  = (const float*)d_in[pbase[p] + 5];
        const float* b2  = (const float*)d_in[pbase[p] + 7];
        gemm(zb, wt_a[p], nullptr, mhb, Nn, 64, 128, b1, bg, bb2, bm, bv, 1);
        gemm(mhb, wt_b[p], out + ooff[p], nullptr, Nn, 128, pout[p],
             b2, nullptr, nullptr, nullptr, nullptr, 0);
    }
}
